// mLSTMMoEBlock_65335042507337
// MI455X (gfx1250) — matrix operand order repaired and checked by reading, not compiled, not run on hardware
//
#include <hip/hip_runtime.h>

// ---------------------------------------------------------------------------
// Types for CDNA5 WMMA
// ---------------------------------------------------------------------------
typedef __attribute__((ext_vector_type(16))) __bf16 v16bf;
typedef __attribute__((ext_vector_type(8)))  float  v8f;

// ---------------------------------------------------------------------------
// Problem constants (match reference)
// ---------------------------------------------------------------------------
constexpr int Bb = 8, Ll = 4096, Dd = 768, Ee = 8, Hh = 1536;
constexpr int T  = Bb * Ll;                 // 32768 tokens
constexpr int TM = 80;                      // tokens per block tile (5 M-tiles)
constexpr int MT = TM / 16;                 // 5 M-tiles
constexpr int NTHREADS = 512;               // 16 waves
constexpr int TILES_PER_E = (T + TM - 1) / TM;   // 410
constexpr int HCHUNK = 128;                 // H columns per phase1/phase2 round

constexpr int XS_STRIDE = Dd + 8;           // 776  (bf16 elems, padded)
constexpr int CS_STRIDE = HCHUNK + 4;       // 132  (fp32 stage, padded)
constexpr int HS_STRIDE = HCHUNK + 8;       // 136  (bf16 elems, padded)

// LDS carve (bytes)
constexpr int LDS_XS  = TM * XS_STRIDE * 2;        // 124160
constexpr int LDS_CS  = TM * CS_STRIDE * 4;        // 42240 (x2: C1s, Cgs)
constexpr int LDS_HS  = TM * HS_STRIDE * 2;        // 21760
constexpr int LDS_TOK = TM * 4;                    // 320
constexpr int LDS_TOTAL = LDS_XS + 2 * LDS_CS + LDS_HS + 2 * LDS_TOK; // 231040

// Workspace layout (bytes)
constexpr unsigned NWF    = (unsigned)Ee * (Hh / 16) * (Dd / 32);   // 18432 frags/matrix
constexpr size_t   WELEMS = (size_t)NWF * 512;                      // bf16 elems/matrix
constexpr size_t OFF_CNT   = 0;                                     // 16 ints
constexpr size_t OFF_LISTS = 256;                                   // int[2][E][T]
constexpr size_t OFF_WTS   = OFF_LISTS + (size_t)2 * Ee * T * 4;    // float[2][T]
constexpr size_t OFF_W1T   = OFF_WTS + (size_t)2 * T * 4;
constexpr size_t OFF_WGT   = OFF_W1T + WELEMS * 2;
constexpr size_t OFF_W2T   = OFF_WGT + WELEMS * 2;

// ---------------------------------------------------------------------------
// Helpers
// ---------------------------------------------------------------------------
static __device__ inline unsigned short f2bf(float f) {
    unsigned int u = __float_as_uint(f);
    unsigned int r = u + 0x7FFFu + ((u >> 16) & 1u);   // round-to-nearest-even
    if ((u & 0x7F800000u) == 0x7F800000u) r = u;       // Inf/NaN: truncate
    return (unsigned short)(r >> 16);
}

static __device__ inline v16bf ld_frag2(const unsigned short* p0,
                                        const unsigned short* p1) {
    union { uint4 u[2]; v16bf v; } t;
    t.u[0] = *reinterpret_cast<const uint4*>(p0);
    t.u[1] = *reinterpret_cast<const uint4*>(p1);
    return t.v;
}

static __device__ inline v8f wmma_bf16(v16bf a, v16bf b, v8f c) {
    return __builtin_amdgcn_wmma_f32_16x16x32_bf16(
        false, a, false, b, (short)0, c, false, false);
}

// ---------------------------------------------------------------------------
// Kernel 0: zero the 2x8 per-(slot,expert) counters
// ---------------------------------------------------------------------------
__global__ void zero_counts_kernel(int* __restrict__ cnt) {
    if (threadIdx.x < 2 * Ee) cnt[threadIdx.x] = 0;
}

// ---------------------------------------------------------------------------
// Kernel 1: router — top-2 + softmax; append token to per-(slot,expert) list.
// List order is nondeterministic (atomics) but every per-token result is
// order-independent, so the final output is deterministic.
// ---------------------------------------------------------------------------
__global__ void router_kernel(const float* __restrict__ scores,
                              int* __restrict__ cnt,
                              int* __restrict__ lists,
                              float* __restrict__ wts) {
    int t = blockIdx.x * blockDim.x + threadIdx.x;
    if (t >= T) return;
    const float* s = scores + (size_t)t * Ee;
    float v[Ee];
#pragma unroll
    for (int i = 0; i < Ee; ++i) v[i] = s[i];
    int i0 = 0; float b0 = v[0];
#pragma unroll
    for (int i = 1; i < Ee; ++i) if (v[i] > b0) { b0 = v[i]; i0 = i; }
    int i1 = -1; float b1 = -3.4e38f;
#pragma unroll
    for (int i = 0; i < Ee; ++i) if (i != i0 && v[i] > b1) { b1 = v[i]; i1 = i; }
    float e1  = __expf(b1 - b0);
    float inv = 1.0f / (1.0f + e1);

    int p0 = atomicAdd(&cnt[0 * Ee + i0], 1);
    lists[((size_t)0 * Ee + i0) * T + p0] = t;
    wts[(size_t)0 * T + t] = inv;

    int p1 = atomicAdd(&cnt[1 * Ee + i1], 1);
    lists[((size_t)1 * Ee + i1) * T + p1] = t;
    wts[(size_t)1 * T + t] = e1 * inv;
}

// ---------------------------------------------------------------------------
// Kernel 2: convert fp32 weights -> bf16 in WMMA B-fragment tiled layout.
// Fragment = 32(K) x 16(N) tile; per lane 16 contiguous bf16 (32 B):
// element i of lane l is W[k = kbase + i + 16*(l/16)][n = nbase + l%16].
// ---------------------------------------------------------------------------
__global__ void convert_weights_kernel(const float* __restrict__ W1,
                                       const float* __restrict__ Wg,
                                       const float* __restrict__ W2,
                                       unsigned short* __restrict__ W1t,
                                       unsigned short* __restrict__ Wgt,
                                       unsigned short* __restrict__ W2t) {
    unsigned g    = blockIdx.x * blockDim.x + threadIdx.x;
    unsigned frag = g >> 5;
    unsigned lane = g & 31u;
    if (frag >= 3u * NWF) return;
    const int lm = lane & 15, lh = lane >> 4;

    unsigned short tmp[16];
    if (frag < 2u * NWF) {                       // W1 or Wg  (D x H per expert)
        const float* Wsrc = (frag < NWF) ? W1 : Wg;
        unsigned short* Wdst = (frag < NWF) ? W1t : Wgt;
        unsigned fi  = (frag < NWF) ? frag : (frag - NWF);
        unsigned e   = fi / ((Hh / 16) * (Dd / 32));
        unsigned rem = fi % ((Hh / 16) * (Dd / 32));
        unsigned ht  = rem / (Dd / 32);          // N tile (H)
        unsigned kt  = rem % (Dd / 32);          // K tile (D)
        const float* src = Wsrc + ((size_t)e * Dd + kt * 32 + lh * 16) * Hh
                                + ht * 16 + lm;
#pragma unroll
        for (int i = 0; i < 16; ++i) tmp[i] = f2bf(src[(size_t)i * Hh]);
        unsigned short* dst = Wdst + ((size_t)fi * 32 + lane) * 16;
        union { unsigned short s[16]; uint4 u[2]; } x;
#pragma unroll
        for (int i = 0; i < 16; ++i) x.s[i] = tmp[i];
        *reinterpret_cast<uint4*>(dst)     = x.u[0];
        *reinterpret_cast<uint4*>(dst + 8) = x.u[1];
    } else {                                     // W2 (H x D per expert)
        unsigned fi  = frag - 2u * NWF;
        unsigned e   = fi / ((Dd / 16) * (Hh / 32));
        unsigned rem = fi % ((Dd / 16) * (Hh / 32));
        unsigned dt  = rem / (Hh / 32);          // N tile (D)
        unsigned ht  = rem % (Hh / 32);          // K tile (H)
        const float* src = W2 + ((size_t)e * Hh + ht * 32 + lh * 16) * Dd
                              + dt * 16 + lm;
#pragma unroll
        for (int i = 0; i < 16; ++i) tmp[i] = f2bf(src[(size_t)i * Dd]);
        unsigned short* dst = W2t + ((size_t)fi * 32 + lane) * 16;
        union { unsigned short s[16]; uint4 u[2]; } x;
#pragma unroll
        for (int i = 0; i < 16; ++i) x.s[i] = tmp[i];
        *reinterpret_cast<uint4*>(dst)     = x.u[0];
        *reinterpret_cast<uint4*>(dst + 8) = x.u[1];
    }
}

// ---------------------------------------------------------------------------
// Kernel 3: routed expert pass. One block = one expert x 80 gathered tokens.
// 512 threads = 16 waves. Per 128-col H chunk:
//   phase1: waves 0-7 compute X@W1 tile, waves 8-15 compute X@Wg tile
//           (each wave: 5 M-tiles x 1 N-tile) -> fp32 stage in LDS
//   gate  : Hs = bf16( c1 * sigmoid(cg) * slot_weight )   (elementwise)
//   phase2: acc[5][3] += Hs @ W2 chunk rows (wave owns 5 M x 3 N tiles)
// Slot 0 pass writes out rows; slot 1 pass accumulates (exactly 1 FP add
// per element in fixed order -> deterministic).
// A-fragment base pointers are loop-invariant and k-offsets are unrolled
// compile-time constants so fragment fetches use immediate offsets only
// (no per-step address VALU -> no WMMA->VALU WAR hazard NOPs).
// ---------------------------------------------------------------------------
__global__ __launch_bounds__(NTHREADS) void moe_pass_kernel(
        const float* __restrict__ X,
        const int*   __restrict__ cnt,
        const int*   __restrict__ lists,
        const float* __restrict__ wts,
        const unsigned short* __restrict__ W1t,
        const unsigned short* __restrict__ Wgt,
        const unsigned short* __restrict__ W2t,
        float* __restrict__ out,
        int slot, int accumulate) {
    extern __shared__ char smem[];
    unsigned short* Xs = (unsigned short*)smem;                   // [80][776]
    float* C1s = (float*)(smem + LDS_XS);                          // [80][132]
    float* Cgs = (float*)(smem + LDS_XS + LDS_CS);                 // [80][132]
    unsigned short* Hs = (unsigned short*)(smem + LDS_XS + 2*LDS_CS); // [80][136]
    int*   tok = (int*)(smem + LDS_XS + 2*LDS_CS + LDS_HS);
    float* wls = (float*)(smem + LDS_XS + 2*LDS_CS + LDS_HS + LDS_TOK);

    const int e     = blockIdx.x / TILES_PER_E;
    const int tile  = blockIdx.x % TILES_PER_E;
    const int count = cnt[slot * Ee + e];
    const int start = tile * TM;
    if (start >= count) return;                    // uniform early exit
    const int valid = (count - start < TM) ? (count - start) : TM;

    const int tid = threadIdx.x;
    const int w = tid >> 5, l = tid & 31;          // wave 0..15, lane 0..31
    const int lm = l & 15, lh = l >> 4;

    // ---- token list + per-token slot weights into LDS ---------------------
    if (tid < TM) {
        int t = (tid < valid)
                    ? lists[((size_t)slot * Ee + e) * T + start + tid] : 0;
        tok[tid] = t;
        wls[tid] = (tid < valid) ? wts[(size_t)slot * T + t] : 0.0f;
    }
    __syncthreads();

    // ---- gather X rows -> Xs bf16 -----------------------------------------
    for (int i = tid; i < TM * Dd / 4; i += NTHREADS) {
        int row = i / (Dd / 4);
        int c4  = i - row * (Dd / 4);
        float4 v = reinterpret_cast<const float4*>(
                       X + (size_t)tok[row] * Dd)[c4];
        unsigned short* p = Xs + row * XS_STRIDE + c4 * 4;
        p[0] = f2bf(v.x); p[1] = f2bf(v.y); p[2] = f2bf(v.z); p[3] = f2bf(v.w);
    }

    // expert weight bases (tiled bf16)
    const unsigned short* WupB =
        ((w < 8) ? W1t : Wgt) + (size_t)e * (Hh / 16) * (Dd / 32) * 512;
    const unsigned short* W2e = W2t + (size_t)e * (Dd / 16) * (Hh / 32) * 512;

    // loop-invariant A-fragment base pointers (one per M-tile row group)
    const unsigned short* apX[MT];
    const unsigned short* apH[MT];
#pragma unroll
    for (int mt = 0; mt < MT; ++mt) {
        apX[mt] = Xs + (mt * 16 + lm) * XS_STRIDE + lh * 16;
        apH[mt] = Hs + (mt * 16 + lm) * HS_STRIDE + lh * 16;
    }

    const v8f vzero = {0.f, 0.f, 0.f, 0.f, 0.f, 0.f, 0.f, 0.f};
    v8f acc[MT][3];                                 // [m-tile][n-group]
#pragma unroll
    for (int mt = 0; mt < MT; ++mt)
#pragma unroll
        for (int j = 0; j < 3; ++j) acc[mt][j] = vzero;

    const int nt1 = w & 7;                          // phase-1 chunk N tile
    for (int hc = 0; hc < Hh; hc += HCHUNK) {
        __syncthreads();   // prev chunk's phase2 reads / gate writes complete

        // ---- phase 1: this wave's up-projection matrix --------------------
        v8f c[MT];
#pragma unroll
        for (int mt = 0; mt < MT; ++mt) c[mt] = vzero;
        const unsigned short* bp =
            WupB + ((size_t)((hc >> 4) + nt1) * (Dd / 32)) * 512
                 + (size_t)l * 16;
#pragma unroll
        for (int kk = 0; kk < Dd; kk += 32) {       // fully unrolled: const kk
            v16bf a[MT];
#pragma unroll
            for (int mt = 0; mt < MT; ++mt)
                a[mt] = ld_frag2(apX[mt] + kk, apX[mt] + kk + 8);
            v16bf bfrag = ld_frag2(bp + (kk >> 5) * 512,
                                   bp + (kk >> 5) * 512 + 8);
#pragma unroll
            for (int mt = 0; mt < MT; ++mt)
                c[mt] = wmma_bf16(a[mt], bfrag, c[mt]);
        }
        float* Cs = (w < 8) ? C1s : Cgs;
        const int col = nt1 * 16 + lm;
#pragma unroll
        for (int mt = 0; mt < MT; ++mt)
#pragma unroll
            for (int r = 0; r < 8; ++r)
                Cs[(mt * 16 + lh * 8 + r) * CS_STRIDE + col] = c[mt][r];
        __syncthreads();

        // ---- gate + weight scale -> Hs bf16 -------------------------------
        for (int i = tid; i < TM * HCHUNK; i += NTHREADS) {
            int m = i >> 7, cc = i & (HCHUNK - 1);
            float v1 = C1s[m * CS_STRIDE + cc];
            float vg = Cgs[m * CS_STRIDE + cc];
            float sg = 1.0f / (1.0f + __expf(-vg));
            Hs[m * HS_STRIDE + cc] = f2bf(v1 * sg * wls[m]);
        }
        __syncthreads();

        // ---- phase 2: acc += Hs @ W2 rows [hc, hc+128) --------------------
        const unsigned short* bp2 =
            W2e + ((size_t)(w * 3) * (Hh / 32) + (hc >> 5)) * 512
                + (size_t)l * 16;
#pragma unroll
        for (int kk = 0; kk < HCHUNK; kk += 32) {   // fully unrolled: const kk
            v16bf a[MT];
#pragma unroll
            for (int mt = 0; mt < MT; ++mt)
                a[mt] = ld_frag2(apH[mt] + kk, apH[mt] + kk + 8);
#pragma unroll
            for (int j = 0; j < 3; ++j) {
                const int fo = (j * (Hh / 32) + (kk >> 5)) * 512;
                v16bf b = ld_frag2(bp2 + fo, bp2 + fo + 8);
#pragma unroll
                for (int mt = 0; mt < MT; ++mt)
                    acc[mt][j] = wmma_bf16(a[mt], b, acc[mt][j]);
            }
        }
    }

    // ---- scatter out rows (masked to valid tokens) ------------------------
#pragma unroll
    for (int mt = 0; mt < MT; ++mt) {
#pragma unroll
        for (int r = 0; r < 8; ++r) {
            const int i = mt * 16 + lh * 8 + r;
            if (i < valid) {
                const int t = tok[i];
#pragma unroll
                for (int j = 0; j < 3; ++j) {
                    const int d = (w * 3 + j) * 16 + lm;
                    float* po = out + (size_t)t * Dd + d;
                    float v = acc[mt][j][r];
                    if (accumulate) *po += v; else *po = v;
                }
            }
        }
    }
}

// ---------------------------------------------------------------------------
// Launch
// ---------------------------------------------------------------------------
extern "C" void kernel_launch(void* const* d_in, const int* in_sizes, int n_in,
                              void* d_out, int out_size, void* d_ws, size_t ws_size,
                              hipStream_t stream) {
    const float* X  = (const float*)d_in[0];   // (B,L,D)
    const float* Sc = (const float*)d_in[1];   // (B,L,E)
    const float* W1 = (const float*)d_in[2];   // (E,D,H)
    const float* Wg = (const float*)d_in[3];   // (E,D,H)
    const float* W2 = (const float*)d_in[4];   // (E,H,D)
    float* out = (float*)d_out;

    char* ws = (char*)d_ws;
    int*   cnt   = (int*)(ws + OFF_CNT);
    int*   lists = (int*)(ws + OFF_LISTS);
    float* wts   = (float*)(ws + OFF_WTS);
    unsigned short* W1t = (unsigned short*)(ws + OFF_W1T);
    unsigned short* Wgt = (unsigned short*)(ws + OFF_WGT);
    unsigned short* W2t = (unsigned short*)(ws + OFF_W2T);

    zero_counts_kernel<<<1, 32, 0, stream>>>(cnt);
    router_kernel<<<T / 256, 256, 0, stream>>>(Sc, cnt, lists, wts);

    const unsigned convThreads = 3u * NWF * 32u;
    convert_weights_kernel<<<convThreads / 256, 256, 0, stream>>>(
        W1, Wg, W2, W1t, Wgt, W2t);

    const int grid = Ee * TILES_PER_E;             // 3280 blocks per pass
    moe_pass_kernel<<<grid, NTHREADS, LDS_TOTAL, stream>>>(
        X, cnt, lists, wts, W1t, Wgt, W2t, out, /*slot=*/0, /*accumulate=*/0);
    moe_pass_kernel<<<grid, NTHREADS, LDS_TOTAL, stream>>>(
        X, cnt, lists, wts, W1t, Wgt, W2t, out, /*slot=*/1, /*accumulate=*/1);
}
